// SimpleGNN_76175539962204
// MI455X (gfx1250) — matrix operand order repaired and checked by reading, not compiled, not run on hardware
//
#include <hip/hip_runtime.h>
#include <hip/hip_bf16.h>
#include <math.h>

#define NNODES 100000
#define FIN    512
#define H1     8
#define C1     8
#define HC     64      // H1*C1
#define NC     16
#define NEG_SLOPE 0.2f

typedef __attribute__((ext_vector_type(16))) __bf16 v16bf;
typedef __attribute__((ext_vector_type(8)))  float  v8f;

// ---------------- helpers ----------------

__device__ __forceinline__ int fkey(float f) {
    int i = __float_as_int(f);
    return i >= 0 ? i : (i ^ 0x7fffffff);   // monotonic float->int key
}
__device__ __forceinline__ float fdec(int k) {
    return __int_as_float(k >= 0 ? k : (k ^ 0x7fffffff));
}
__device__ __forceinline__ void edge_sd(const int* __restrict__ ei, int E, int e,
                                        int& s, int& d) {
    if (e < E) { s = ei[e]; d = ei[E + e]; }
    else       { s = e - E; d = e - E; }     // self-loops appended
}
__device__ __forceinline__ v16bf cvt_a_frag(float4 p0, float4 p1, float4 q0, float4 q1) {
    v16bf a;
    a[0]  = (__bf16)p0.x; a[1]  = (__bf16)p0.y; a[2]  = (__bf16)p0.z; a[3]  = (__bf16)p0.w;
    a[4]  = (__bf16)p1.x; a[5]  = (__bf16)p1.y; a[6]  = (__bf16)p1.z; a[7]  = (__bf16)p1.w;
    a[8]  = (__bf16)q0.x; a[9]  = (__bf16)q0.y; a[10] = (__bf16)q0.z; a[11] = (__bf16)q0.w;
    a[12] = (__bf16)q1.x; a[13] = (__bf16)q1.y; a[14] = (__bf16)q1.z; a[15] = (__bf16)q1.w;
    return a;
}

// ---------------- weight conversion (fp32 -> bf16, transposed) ----------------

__global__ void k_convert_weights(const float* __restrict__ W1,
                                  const float* __restrict__ W2,
                                  __bf16* __restrict__ w1t,   // [HC x FIN]
                                  __bf16* __restrict__ w2t) { // [NC x HC]
    int idx = blockIdx.x * blockDim.x + threadIdx.x;
    if (idx < FIN * HC) {
        int k = idx / HC, n = idx % HC;
        w1t[(size_t)n * FIN + k] = (__bf16)W1[idx];
    }
    if (idx < HC * NC) {
        int k = idx / NC, n = idx % NC;
        w2t[(size_t)n * HC + k] = (__bf16)W2[idx];
    }
}

// ---------------- per-call scratch init ----------------

__global__ void k_init(float* __restrict__ out1, float* __restrict__ out2,
                       float* __restrict__ den1, float* __restrict__ den2,
                       int*   __restrict__ max1, int*   __restrict__ max2) {
    int idx = blockIdx.x * blockDim.x + threadIdx.x;
    if (idx < NNODES * HC) out1[idx] = 0.0f;
    if (idx < NNODES * NC) out2[idx] = 0.0f;
    if (idx < NNODES * H1) { den1[idx] = 0.0f; max1[idx] = (int)0x80000000; }
    if (idx < NNODES)      { den2[idx] = 0.0f; max2[idx] = (int)0x80000000; }
}

// ---------------- GEMM1: h1 = x @ W1  (WMMA bf16, fp32 accumulate) ----------------
// B (W1 bf16) staged into LDS in WMMA-fragment order:
//   lB[((k0/32)*4 + t)*32 + lane][16] ; each lane reads one aligned 32B chunk.

__global__ __launch_bounds__(256)
void k_gemm1(const float* __restrict__ x, const __bf16* __restrict__ w1t,
             float* __restrict__ h1) {
    __shared__ __bf16 lB[16 * 4 * 32 * 16];          // 64 KB, exactly W1 in bf16

    const int tid = threadIdx.x;

    // ---- cooperative stage: 2048 groups of 32B, 8 per thread ----
    #pragma unroll
    for (int i = 0; i < 8; ++i) {
        int g    = (i << 8) + tid;                   // 0..2047
        int k0s  = g >> 7;                           // k-step (k0/32)
        int rem  = g & 127;
        int t    = rem >> 5;                         // N-tile
        int ln   = rem & 31;                         // lane slot
        int col  = (t << 4) + (ln & 15);
        int koff = (k0s << 5) + ((ln & 16) ? 8 : 0);
        const uint4* src = (const uint4*)(w1t + (size_t)col * FIN + koff);
        uint4* dst = (uint4*)(&lB[g << 4]);
        dst[0] = src[0];
        dst[1] = src[2];
    }
    __syncthreads();

    const int wave = tid >> 5;
    const int lane = tid & 31;
    const int ntiles = (NNODES + 15) >> 4;
    const int tile = blockIdx.x * 8 + wave;

    if (tile < ntiles) {
        const int rowBase = tile << 4;
        int rowg = rowBase + (lane & 15);
        if (rowg >= NNODES) rowg = NNODES - 1;       // duplicate row in tail tile
        const int kb  = (lane >> 4) << 3;            // A K-half: 0 or 8
        const int col = lane & 15;
        const float* xrow = x + (size_t)rowg * FIN;

        v8f acc0 = {}, acc1 = {}, acc2 = {}, acc3 = {};

        // pipeline prologue: A chunk for k0 = 0
        const float* xr0 = xrow + kb;
        float4 p0 = *(const float4*)(xr0);
        float4 p1 = *(const float4*)(xr0 + 4);
        float4 q0 = *(const float4*)(xr0 + 16);
        float4 q1 = *(const float4*)(xr0 + 20);

        for (int k0 = 0; k0 < FIN; k0 += 32) {
            v16bf a = cvt_a_frag(p0, p1, q0, q1);

            if (k0 + 32 < FIN) {                     // issue next A chunk early
                const float* xn = xrow + k0 + 32 + kb;
                if (k0 + 64 < FIN) __builtin_prefetch(xn + 32, 0, 3);
                p0 = *(const float4*)(xn);
                p1 = *(const float4*)(xn + 4);
                q0 = *(const float4*)(xn + 16);
                q1 = *(const float4*)(xn + 20);
            }

            // B fragments from LDS: base = (k0/32)*2048 + lane*16, tile stride 512
            const v16bf* fb = (const v16bf*)(&lB[((k0 >> 5) << 11) + (lane << 4)]);
            v16bf b0 = fb[0];
            v16bf b1 = fb[32];
            v16bf b2 = fb[64];
            v16bf b3 = fb[96];

            acc0 = __builtin_amdgcn_wmma_f32_16x16x32_bf16(false, a, false, b0, (short)0, acc0, false, false);
            acc1 = __builtin_amdgcn_wmma_f32_16x16x32_bf16(false, a, false, b1, (short)0, acc1, false, false);
            acc2 = __builtin_amdgcn_wmma_f32_16x16x32_bf16(false, a, false, b2, (short)0, acc2, false, false);
            acc3 = __builtin_amdgcn_wmma_f32_16x16x32_bf16(false, a, false, b3, (short)0, acc3, false, false);
        }

        const int rsel = (lane >> 4) << 3;           // C/D: rows 0-7 or 8-15
        #pragma unroll
        for (int j = 0; j < 8; ++j) {
            int node = rowBase + j + rsel;
            if (node < NNODES) {
                float* hr = h1 + (size_t)node * HC + col;
                hr[0]  = acc0[j];
                hr[16] = acc1[j];
                hr[32] = acc2[j];
                hr[48] = acc3[j];
            }
        }
    }
}

// ---------------- attention coefficients, layer 1 ----------------

__global__ void k_att1(const float* __restrict__ h1,
                       const float* __restrict__ a1s, const float* __restrict__ a1d,
                       float* __restrict__ al1s, float* __restrict__ al1d) {
    int idx = blockIdx.x * blockDim.x + threadIdx.x;       // n*H1 + h
    if (idx >= NNODES * H1) return;
    int n = idx / H1, h = idx % H1;
    const float* hp = h1 + (size_t)n * HC + h * C1;
    float s = 0.0f, d = 0.0f;
    #pragma unroll
    for (int c = 0; c < C1; ++c) {
        float v = hp[c];
        s += v * a1s[h * C1 + c];
        d += v * a1d[h * C1 + c];
    }
    al1s[idx] = s;
    al1d[idx] = d;
}

// ---------------- layer-1 edge passes ----------------

__global__ void k_edge1_max(const int* __restrict__ ei, int E, int ne,
                            const float* __restrict__ al1s, const float* __restrict__ al1d,
                            int* __restrict__ max1) {
    int idx = blockIdx.x * blockDim.x + threadIdx.x;       // e*H1 + h
    if (idx >= ne * H1) return;
    int e = idx / H1, h = idx % H1, s, d;
    edge_sd(ei, E, e, s, d);
    float v = al1s[s * H1 + h] + al1d[d * H1 + h];
    v = v > 0.0f ? v : NEG_SLOPE * v;
    atomicMax(&max1[d * H1 + h], fkey(v));
}

__global__ void k_edge1_sum(const int* __restrict__ ei, int E, int ne,
                            const float* __restrict__ al1s, const float* __restrict__ al1d,
                            const int* __restrict__ max1, float* __restrict__ den1) {
    int idx = blockIdx.x * blockDim.x + threadIdx.x;
    if (idx >= ne * H1) return;
    int e = idx / H1, h = idx % H1, s, d;
    edge_sd(ei, E, e, s, d);
    float v = al1s[s * H1 + h] + al1d[d * H1 + h];
    v = v > 0.0f ? v : NEG_SLOPE * v;
    float m = fdec(max1[d * H1 + h]);
    atomicAdd(&den1[d * H1 + h], __expf(v - m));
}

__global__ void k_edge1_agg(const int* __restrict__ ei, int E, int ne,
                            const float* __restrict__ al1s, const float* __restrict__ al1d,
                            const int* __restrict__ max1, const float* __restrict__ den1,
                            const float* __restrict__ h1, float* __restrict__ out1) {
    int idx = blockIdx.x * blockDim.x + threadIdx.x;
    if (idx >= ne * H1) return;
    int e = idx / H1, h = idx % H1, s, d;
    edge_sd(ei, E, e, s, d);
    float v = al1s[s * H1 + h] + al1d[d * H1 + h];
    v = v > 0.0f ? v : NEG_SLOPE * v;
    float m = fdec(max1[d * H1 + h]);
    float alpha = __expf(v - m) / den1[d * H1 + h];
    const float* hs = h1 + (size_t)s * HC + h * C1;
    float* od = out1 + (size_t)d * HC + h * C1;
    #pragma unroll
    for (int c = 0; c < C1; ++c)
        atomicAdd(&od[c], alpha * hs[c]);
}

// ---------------- bias + ELU ----------------

__global__ void k_elu(const float* __restrict__ out1, const float* __restrict__ b1,
                      float* __restrict__ h2in) {
    int idx = blockIdx.x * blockDim.x + threadIdx.x;
    if (idx >= NNODES * HC) return;
    float v = out1[idx] + b1[idx % HC];
    h2in[idx] = v > 0.0f ? v : (__expf(v) - 1.0f);
}

// ---------------- GEMM2: h2l = h2in @ W2 (WMMA, LDS-staged B) ----------------

__global__ __launch_bounds__(256)
void k_gemm2(const float* __restrict__ h2in, const __bf16* __restrict__ w2t,
             float* __restrict__ h2l) {
    __shared__ __bf16 lB[2 * 32 * 16];               // 2 KB, W2 in fragment order

    const int tid = threadIdx.x;
    if (tid < 64) {
        int k0s  = tid >> 5;
        int ln   = tid & 31;
        int col  = ln & 15;
        int koff = (k0s << 5) + ((ln & 16) ? 8 : 0);
        const uint4* src = (const uint4*)(w2t + (size_t)col * HC + koff);
        uint4* dst = (uint4*)(&lB[tid << 4]);
        dst[0] = src[0];
        dst[1] = src[2];
    }
    __syncthreads();

    const int wave = tid >> 5;
    const int lane = tid & 31;
    const int ntiles = (NNODES + 15) >> 4;
    const int tile = blockIdx.x * 8 + wave;

    if (tile < ntiles) {
        const int rowBase = tile << 4;
        int rowg = rowBase + (lane & 15);
        if (rowg >= NNODES) rowg = NNODES - 1;
        const int kb  = (lane >> 4) << 3;
        const int col = lane & 15;

        v8f acc = {};
        const float* xrow = h2in + (size_t)rowg * HC;

        #pragma unroll
        for (int k0 = 0; k0 < HC; k0 += 32) {
            const float* xr = xrow + k0 + kb;
            float4 p0 = *(const float4*)(xr);
            float4 p1 = *(const float4*)(xr + 4);
            float4 q0 = *(const float4*)(xr + 16);
            float4 q1 = *(const float4*)(xr + 20);
            v16bf a = cvt_a_frag(p0, p1, q0, q1);

            v16bf b = *(const v16bf*)(&lB[((k0 >> 5) << 9) + (lane << 4)]);
            acc = __builtin_amdgcn_wmma_f32_16x16x32_bf16(false, a, false, b, (short)0, acc, false, false);
        }

        const int rsel = (lane >> 4) << 3;
        #pragma unroll
        for (int j = 0; j < 8; ++j) {
            int node = rowBase + j + rsel;
            if (node < NNODES)
                h2l[(size_t)node * NC + col] = acc[j];
        }
    }
}

// ---------------- attention coefficients, layer 2 ----------------

__global__ void k_att2(const float* __restrict__ h2l,
                       const float* __restrict__ a2s, const float* __restrict__ a2d,
                       float* __restrict__ al2s, float* __restrict__ al2d) {
    int n = blockIdx.x * blockDim.x + threadIdx.x;
    if (n >= NNODES) return;
    const float* hp = h2l + (size_t)n * NC;
    float s = 0.0f, d = 0.0f;
    #pragma unroll
    for (int c = 0; c < NC; ++c) {
        float v = hp[c];
        s += v * a2s[c];
        d += v * a2d[c];
    }
    al2s[n] = s;
    al2d[n] = d;
}

// ---------------- layer-2 edge passes (H=1, C=16) ----------------

__global__ void k_edge2_max(const int* __restrict__ ei, int E, int ne,
                            const float* __restrict__ al2s, const float* __restrict__ al2d,
                            int* __restrict__ max2) {
    int e = blockIdx.x * blockDim.x + threadIdx.x;
    if (e >= ne) return;
    int s, d; edge_sd(ei, E, e, s, d);
    float v = al2s[s] + al2d[d];
    v = v > 0.0f ? v : NEG_SLOPE * v;
    atomicMax(&max2[d], fkey(v));
}

__global__ void k_edge2_sum(const int* __restrict__ ei, int E, int ne,
                            const float* __restrict__ al2s, const float* __restrict__ al2d,
                            const int* __restrict__ max2, float* __restrict__ den2) {
    int e = blockIdx.x * blockDim.x + threadIdx.x;
    if (e >= ne) return;
    int s, d; edge_sd(ei, E, e, s, d);
    float v = al2s[s] + al2d[d];
    v = v > 0.0f ? v : NEG_SLOPE * v;
    atomicAdd(&den2[d], __expf(v - fdec(max2[d])));
}

__global__ void k_edge2_agg(const int* __restrict__ ei, int E, int ne,
                            const float* __restrict__ al2s, const float* __restrict__ al2d,
                            const int* __restrict__ max2, const float* __restrict__ den2,
                            const float* __restrict__ h2l, float* __restrict__ out2) {
    int e = blockIdx.x * blockDim.x + threadIdx.x;
    if (e >= ne) return;
    int s, d; edge_sd(ei, E, e, s, d);
    float v = al2s[s] + al2d[d];
    v = v > 0.0f ? v : NEG_SLOPE * v;
    float alpha = __expf(v - fdec(max2[d])) / den2[d];
    const float* hs = h2l + (size_t)s * NC;
    float* od = out2 + (size_t)d * NC;
    #pragma unroll
    for (int c = 0; c < NC; ++c)
        atomicAdd(&od[c], alpha * hs[c]);
}

// ---------------- bias + log_softmax ----------------

__global__ void k_final(const float* __restrict__ out2, const float* __restrict__ b2,
                        float* __restrict__ y) {
    int n = blockIdx.x * blockDim.x + threadIdx.x;
    if (n >= NNODES) return;
    float v[NC];
    float m = -3.402823466e38f;
    #pragma unroll
    for (int c = 0; c < NC; ++c) {
        v[c] = out2[(size_t)n * NC + c] + b2[c];
        m = fmaxf(m, v[c]);
    }
    float s = 0.0f;
    #pragma unroll
    for (int c = 0; c < NC; ++c) s += __expf(v[c] - m);
    float ls = __logf(s);
    #pragma unroll
    for (int c = 0; c < NC; ++c)
        y[(size_t)n * NC + c] = v[c] - m - ls;
}

// ---------------- launcher ----------------

extern "C" void kernel_launch(void* const* d_in, const int* in_sizes, int n_in,
                              void* d_out, int out_size, void* d_ws, size_t ws_size,
                              hipStream_t stream) {
    const float* x    = (const float*)d_in[0];
    const int*   ei   = (const int*)  d_in[1];
    const float* W1   = (const float*)d_in[2];
    const float* a1s  = (const float*)d_in[3];
    const float* a1d  = (const float*)d_in[4];
    const float* b1   = (const float*)d_in[5];
    const float* W2   = (const float*)d_in[6];
    const float* a2s  = (const float*)d_in[7];
    const float* a2d  = (const float*)d_in[8];
    const float* b2   = (const float*)d_in[9];
    float* y = (float*)d_out;

    const int E  = in_sizes[1] / 2;
    const int ne = E + NNODES;

    // workspace carve-out (256B aligned)
    char* wp = (char*)d_ws;
    auto carve = [&](size_t bytes) {
        void* p = (void*)wp;
        wp += (bytes + 255) & ~(size_t)255;
        return p;
    };
    __bf16* w1t  = (__bf16*)carve((size_t)FIN * HC * 2);
    __bf16* w2t  = (__bf16*)carve((size_t)HC * NC * 2);
    float*  h1   = (float*) carve((size_t)NNODES * HC * 4);   // reused as h2in
    float*  al1s = (float*) carve((size_t)NNODES * H1 * 4);
    float*  al1d = (float*) carve((size_t)NNODES * H1 * 4);
    int*    max1 = (int*)   carve((size_t)NNODES * H1 * 4);
    float*  den1 = (float*) carve((size_t)NNODES * H1 * 4);
    float*  out1 = (float*) carve((size_t)NNODES * HC * 4);
    float*  h2l  = (float*) carve((size_t)NNODES * NC * 4);
    float*  al2s = (float*) carve((size_t)NNODES * 4);
    float*  al2d = (float*) carve((size_t)NNODES * 4);
    int*    max2 = (int*)   carve((size_t)NNODES * 4);
    float*  den2 = (float*) carve((size_t)NNODES * 4);
    float*  out2 = (float*) carve((size_t)NNODES * NC * 4);

    const int T = 256;
    auto nb = [](long long n, int t) { return (int)((n + t - 1) / t); };
    const int gemmBlocks = nb(((NNODES + 15) / 16), 8);       // 8 waves/block, 1 tile/wave

    k_convert_weights<<<nb(FIN * HC, T), T, 0, stream>>>(W1, W2, w1t, w2t);
    k_init<<<nb((long long)NNODES * HC, T), T, 0, stream>>>(out1, out2, den1, den2, max1, max2);

    k_gemm1<<<gemmBlocks, T, 0, stream>>>(x, w1t, h1);
    k_att1<<<nb((long long)NNODES * H1, T), T, 0, stream>>>(h1, a1s, a1d, al1s, al1d);

    k_edge1_max<<<nb((long long)ne * H1, T), T, 0, stream>>>(ei, E, ne, al1s, al1d, max1);
    k_edge1_sum<<<nb((long long)ne * H1, T), T, 0, stream>>>(ei, E, ne, al1s, al1d, max1, den1);
    k_edge1_agg<<<nb((long long)ne * H1, T), T, 0, stream>>>(ei, E, ne, al1s, al1d, max1, den1, h1, out1);

    float* h2in = h1;   // h1 dead after edge1_agg; reuse buffer
    k_elu<<<nb((long long)NNODES * HC, T), T, 0, stream>>>(out1, b1, h2in);

    k_gemm2<<<gemmBlocks, T, 0, stream>>>(h2in, w2t, h2l);
    k_att2<<<nb(NNODES, T), T, 0, stream>>>(h2l, a2s, a2d, al2s, al2d);

    k_edge2_max<<<nb(ne, T), T, 0, stream>>>(ei, E, ne, al2s, al2d, max2);
    k_edge2_sum<<<nb(ne, T), T, 0, stream>>>(ei, E, ne, al2s, al2d, max2, den2);
    k_edge2_agg<<<nb(ne, T), T, 0, stream>>>(ei, E, ne, al2s, al2d, max2, den2, h2l, out2);

    k_final<<<nb(NNODES, T), T, 0, stream>>>(out2, b2, y);
}
